// ForgettingAttentionLayer_57878979281198
// MI455X (gfx1250) — matrix operand order repaired and checked by reading, not compiled, not run on hardware
//
#include <hip/hip_runtime.h>
#include <hip/hip_bf16.h>

typedef __attribute__((ext_vector_type(16))) _Float16 v16h;
typedef __attribute__((ext_vector_type(8)))  float    v8f;

#define T_SEQ   2048
#define D_MODEL 2048
#define NHEAD   32
#define HDIM    64

static __device__ __forceinline__ v8f wmma_f16(const v16h& a, const v16h& b, const v8f& c) {
  // D = A(16x32 f16) * B(32x16 f16) + C(16x16 f32)
  return __builtin_amdgcn_wmma_f32_16x16x32_f16(false, a, false, b, (short)0, c,
                                                false, false);
}

// CDNA5 async copy: global -> LDS, 16B per lane, tracked by ASYNCcnt.
static __device__ __forceinline__ void async_copy_b128(unsigned lds_off, const void* gaddr) {
  asm volatile("global_load_async_to_lds_b128 %0, %1, off"
               :: "v"(lds_off), "v"(gaddr) : "memory");
}
static __device__ __forceinline__ void wait_async0() {
  asm volatile("s_wait_asynccnt 0x0" ::: "memory");
}

// ---------------------------------------------------------------------------
// streaming fp32 -> f16 conversion (one pass per matrix; 2048x2048)
// ---------------------------------------------------------------------------
__global__ __launch_bounds__(256) void cvt_f16_kernel(const float* __restrict__ s,
                                                      _Float16* __restrict__ d) {
  size_t i = ((size_t)blockIdx.x * 256 + threadIdx.x) * 8;
#pragma unroll
  for (int e = 0; e < 8; ++e) d[i + e] = (_Float16)s[i + e];
}

// ---------------------------------------------------------------------------
// C[2048 x 2048] = A * W^T, all-f16 inputs (pre-converted), f32 accumulate.
// Block: 256 thr = 8 waves; block tile 128(M) x 64(N); wave tile 16 x 64
// (4 accumulators -> 4 WMMAs per 32-k step off one A fragment).
// Tiles staged via GLOBAL_LOAD_ASYNC_TO_LDS_B128 with double buffering.
// ---------------------------------------------------------------------------
template <bool OUT_F16>
__global__ __launch_bounds__(256) void gemm_f16_kernel(const _Float16* __restrict__ A,
                                                       const _Float16* __restrict__ W,
                                                       void* __restrict__ Cv) {
  // rows padded to 40 halves (80B): keeps 16B alignment, spreads banks
  __shared__ _Float16 As[2][128][40];
  __shared__ _Float16 Ws[2][64][40];

  const int tid  = threadIdx.x;
  const int lane = tid & 31;
  const int wave = tid >> 5;
  const int row0 = blockIdx.y * 128;
  const int col0 = blockIdx.x * 64;
  const int m_off = wave * 16;
  const int lm    = lane & 15;
  const int abase = (lane < 16) ? 0 : 8;    // A-frag K sub-offset (ISA layout)
  const int bbase = (lane < 16) ? 0 : 16;   // B-frag K range base
  const int hi8   = (lane < 16) ? 0 : 8;    // C/D row offset

  // tile loader: As 128x32 halves (each thread 32B = 2 x b128),
  //              Ws  64x32 halves (each thread 16B = 1 x b128)
  auto load_tiles = [&](int buf, int k0) {
    {
      int r = tid >> 1, c0 = (tid & 1) * 16;
      const _Float16* g = A + (size_t)(row0 + r) * D_MODEL + k0 + c0;
      unsigned l = (unsigned)(unsigned long long)&As[buf][r][c0];
      async_copy_b128(l, g);
      async_copy_b128(l + 16, g + 8);
    }
    {
      int r = tid >> 2, c0 = (tid & 3) * 8;
      const _Float16* g = W + (size_t)(col0 + r) * D_MODEL + k0 + c0;
      async_copy_b128((unsigned)(unsigned long long)&Ws[buf][r][c0], g);
    }
  };

  v8f acc[4];
  const v8f zerov = {0.f, 0.f, 0.f, 0.f, 0.f, 0.f, 0.f, 0.f};
#pragma unroll
  for (int n = 0; n < 4; ++n) acc[n] = zerov;

  load_tiles(0, 0);
  int buf = 0;
  for (int k0 = 0; k0 < D_MODEL; k0 += 32) {
    wait_async0();        // our async copies into `buf` are done
    __syncthreads();      // everyone's copies are done
    if (k0 + 32 < D_MODEL) load_tiles(buf ^ 1, k0 + 32);  // overlap next tile

    // A fragment: lane holds row (m_off+lm), K contiguous bbase..bbase+15
    v16h a;
    const _Float16* arow = &As[buf][m_off + lm][0];
#pragma unroll
    for (int j = 0; j < 8; ++j) {
      int kk = bbase + 2 * j;
      a[2 * j]     = arow[kk];
      a[2 * j + 1] = arow[kk + 1];
    }
    // 4 B fragments -> 4 WMMAs off the same A fragment
#pragma unroll
    for (int n = 0; n < 4; ++n) {
      v16h b;
      const _Float16* brow = &Ws[buf][n * 16 + lm][0];
#pragma unroll
      for (int h = 0; h < 16; ++h) b[h] = brow[bbase + h];
      acc[n] = wmma_f16(a, b, acc[n]);
    }
    __syncthreads();      // all waves done reading `buf` before it is refilled
    buf ^= 1;
  }

#pragma unroll
  for (int n = 0; n < 4; ++n)
#pragma unroll
    for (int r = 0; r < 8; ++r) {
      int gr = row0 + m_off + r + hi8;
      int gc = col0 + n * 16 + lm;
      size_t oi = (size_t)gr * D_MODEL + gc;
      if (OUT_F16)
        ((_Float16*)Cv)[oi] = (_Float16)acc[n][r];
      else
        ((float*)Cv)[oi] = acc[n][r];
    }
}

// ---------------------------------------------------------------------------
// logf[h][t] = log_sigmoid(x[t] . Wf[h] + bf[h])   (fp32, stable form)
// ---------------------------------------------------------------------------
__global__ __launch_bounds__(256) void fgate_kernel(const float* __restrict__ x,
                                                    const float* __restrict__ Wf,
                                                    const float* __restrict__ bf,
                                                    float* __restrict__ logf) {
  int t  = blockIdx.x * 8 + (threadIdx.x >> 5);
  int hh = threadIdx.x & 31;
  const float* xr = x + (size_t)t * D_MODEL;
  const float* wr = Wf + (size_t)hh * D_MODEL;
  float acc = 0.f;
  for (int k = 0; k < D_MODEL; k += 4) {
    acc += xr[k] * wr[k] + xr[k + 1] * wr[k + 1] + xr[k + 2] * wr[k + 2] +
           xr[k + 3] * wr[k + 3];
  }
  float z  = acc + bf[hh];
  float lf = (z >= 0.f) ? -log1pf(__expf(-z)) : (z - log1pf(__expf(z)));
  logf[(size_t)hh * T_SEQ + t] = lf;
}

// per-head inclusive cumsum over T (tiny; 32 serial scans)
__global__ __launch_bounds__(32) void cumsum_kernel(float* __restrict__ c) {
  float* row = c + (size_t)threadIdx.x * T_SEQ;
  float run = 0.f;
  for (int t = 0; t < T_SEQ; ++t) {
    run += row[t];
    row[t] = run;
  }
}

// ---------------------------------------------------------------------------
// Flash attention w/ log-decay bias. One wave per (head, 16 query rows).
// scores = QK^T * 0.125 + c_i - c_j ; causal ; online softmax ; O += P V
// ---------------------------------------------------------------------------
__global__ __launch_bounds__(32) void attn_kernel(const _Float16* __restrict__ qh,
                                                  const _Float16* __restrict__ kh,
                                                  const _Float16* __restrict__ vh,
                                                  const float* __restrict__ cdec,
                                                  _Float16* __restrict__ oh) {
  __shared__ _Float16 Pl[16][34];   // P tile staging (C-layout -> A-layout)

  const int lane = threadIdx.x;
  const int i0   = blockIdx.x * 16;
  const int head = blockIdx.y;
  const int lm    = lane & 15;
  const int hi8   = (lane < 16) ? 0 : 8;
  const int abase = hi8;
  const int bbase = (lane < 16) ? 0 : 16;
  const float scale = 0.125f;   // 1/sqrt(64)

  // Q A-fragments for feature dims [0,32) and [32,64), kept in registers
  v16h a0, a1;
  const _Float16* qrow = qh + (size_t)(i0 + lm) * D_MODEL + head * HDIM;
#pragma unroll
  for (int j = 0; j < 8; ++j) {
    int kk = bbase + 2 * j;
    a0[2 * j]     = qrow[kk];
    a0[2 * j + 1] = qrow[kk + 1];
    a1[2 * j]     = qrow[32 + kk];
    a1[2 * j + 1] = qrow[32 + kk + 1];
  }

  float ci[8];
#pragma unroll
  for (int r = 0; r < 8; ++r)
    ci[r] = cdec[(size_t)head * T_SEQ + i0 + r + hi8];

  float mrow[8], srow[8];
#pragma unroll
  for (int r = 0; r < 8; ++r) { mrow[r] = -3.0e38f; srow[r] = 0.f; }
  const v8f zerov = {0.f, 0.f, 0.f, 0.f, 0.f, 0.f, 0.f, 0.f};
  v8f O[4];
#pragma unroll
  for (int n = 0; n < 4; ++n) O[n] = zerov;

  const int ntiles = (i0 + 16 + 31) >> 5;   // 32-key tiles, causal bound
  for (int jt = 0; jt < ntiles; ++jt) {
    const int j0 = jt * 32;
    float sc[2][8];

#pragma unroll
    for (int s = 0; s < 2; ++s) {
      const int j0s = j0 + s * 16;
      // B fragment of K^T: col n = key (lm), K = feature dim (bbase+h)
      v16h b0, b1;
      const _Float16* krow = kh + (size_t)(j0s + lm) * D_MODEL + head * HDIM;
#pragma unroll
      for (int h = 0; h < 16; ++h) {
        b0[h] = krow[bbase + h];
        b1[h] = krow[32 + bbase + h];
      }
      v8f sacc = zerov;
      sacc = wmma_f16(a0, b0, sacc);
      sacc = wmma_f16(a1, b1, sacc);
      float cj = cdec[(size_t)head * T_SEQ + j0s + lm];
#pragma unroll
      for (int r = 0; r < 8; ++r) {
        int gi = i0 + r + hi8;
        int gj = j0s + lm;
        float v = sacc[r] * scale + ci[r] - cj;
        sc[s][r] = (gj <= gi) ? v : -3.0e38f;
      }
    }

    // online softmax update (row reductions across 16-lane half-wave)
    float corr[8];
#pragma unroll
    for (int r = 0; r < 8; ++r) {
      float rmax = fmaxf(sc[0][r], sc[1][r]);
#pragma unroll
      for (int off = 1; off < 16; off <<= 1)
        rmax = fmaxf(rmax, __shfl_xor(rmax, off, 32));
      float mnew = fmaxf(mrow[r], rmax);
      float p0 = __expf(sc[0][r] - mnew);
      float p1 = __expf(sc[1][r] - mnew);
      float rs = p0 + p1;
#pragma unroll
      for (int off = 1; off < 16; off <<= 1)
        rs += __shfl_xor(rs, off, 32);
      corr[r] = __expf(mrow[r] - mnew);
      srow[r] = srow[r] * corr[r] + rs;
      mrow[r] = mnew;
      Pl[r + hi8][lm]      = (_Float16)p0;   // P in C-layout -> LDS
      Pl[r + hi8][16 + lm] = (_Float16)p1;
    }
#pragma unroll
    for (int n = 0; n < 4; ++n)
#pragma unroll
      for (int r = 0; r < 8; ++r) O[n][r] *= corr[r];

    __syncthreads();   // single wave: orders LDS RAW across lanes

    // re-read P as A fragment (16 rows x 32 keys)
    v16h pa;
#pragma unroll
    for (int j = 0; j < 8; ++j) {
      int kk = bbase + 2 * j;
      pa[2 * j]     = Pl[lm][kk];
      pa[2 * j + 1] = Pl[lm][kk + 1];
    }

    // O[16 x 64] += P[16 x 32] * V[32 x 64]  (four 16-wide N chunks)
#pragma unroll
    for (int n = 0; n < 4; ++n) {
      v16h bv;
#pragma unroll
      for (int h = 0; h < 16; ++h)
        bv[h] = vh[(size_t)(j0 + bbase + h) * D_MODEL + head * HDIM + n * 16 + lm];
      O[n] = wmma_f16(pa, bv, O[n]);
    }
    __syncthreads();
  }

#pragma unroll
  for (int n = 0; n < 4; ++n)
#pragma unroll
    for (int r = 0; r < 8; ++r) {
      float ov = O[n][r] / srow[r];
      oh[(size_t)(i0 + r + hi8) * D_MODEL + head * HDIM + n * 16 + lm] =
          (_Float16)ov;
    }
}

// ---------------------------------------------------------------------------
extern "C" void kernel_launch(void* const* d_in, const int* in_sizes, int n_in,
                              void* d_out, int out_size, void* d_ws, size_t ws_size,
                              hipStream_t stream) {
  (void)in_sizes; (void)n_in; (void)out_size; (void)ws_size;
  const float* x  = (const float*)d_in[0];
  const float* Wq = (const float*)d_in[1];
  const float* Wk = (const float*)d_in[2];
  const float* Wv = (const float*)d_in[3];
  const float* Wf = (const float*)d_in[4];
  const float* bf = (const float*)d_in[5];
  const float* Wo = (const float*)d_in[6];
  float* out = (float*)d_out;

  // workspace: xh (aliased later by attn output), qh, kh, vh,
  //            Wqh, Wkh, Wvh, Woh (f16), cdec (f32 H*T)     ~64.3 MB total
  char* ws = (char*)d_ws;
  const size_t td = (size_t)T_SEQ * D_MODEL;
  _Float16* xh  = (_Float16*)ws; ws += td * sizeof(_Float16);   // later: attn out
  _Float16* qh  = (_Float16*)ws; ws += td * sizeof(_Float16);
  _Float16* kh  = (_Float16*)ws; ws += td * sizeof(_Float16);
  _Float16* vh  = (_Float16*)ws; ws += td * sizeof(_Float16);
  _Float16* Wqh = (_Float16*)ws; ws += td * sizeof(_Float16);
  _Float16* Wkh = (_Float16*)ws; ws += td * sizeof(_Float16);
  _Float16* Wvh = (_Float16*)ws; ws += td * sizeof(_Float16);
  _Float16* Woh = (_Float16*)ws; ws += td * sizeof(_Float16);
  float* cdec   = (float*)ws;                                   // NHEAD*T_SEQ

  const int cvt_blocks = (int)(td / (256 * 8));
  cvt_f16_kernel<<<cvt_blocks, 256, 0, stream>>>(x, xh);
  cvt_f16_kernel<<<cvt_blocks, 256, 0, stream>>>(Wq, Wqh);
  cvt_f16_kernel<<<cvt_blocks, 256, 0, stream>>>(Wk, Wkh);
  cvt_f16_kernel<<<cvt_blocks, 256, 0, stream>>>(Wv, Wvh);
  cvt_f16_kernel<<<cvt_blocks, 256, 0, stream>>>(Wo, Woh);

  dim3 gg(D_MODEL / 64, T_SEQ / 128);
  gemm_f16_kernel<true><<<gg, 256, 0, stream>>>(xh, Wqh, (void*)qh);
  gemm_f16_kernel<true><<<gg, 256, 0, stream>>>(xh, Wkh, (void*)kh);
  gemm_f16_kernel<true><<<gg, 256, 0, stream>>>(xh, Wvh, (void*)vh);

  fgate_kernel<<<T_SEQ / 8, 256, 0, stream>>>(x, Wf, bf, cdec);
  cumsum_kernel<<<1, 32, 0, stream>>>(cdec);

  _Float16* ah = xh;   // xh is dead after the QKV GEMMs; reuse for attn output
  attn_kernel<<<dim3(T_SEQ / 16, NHEAD), 32, 0, stream>>>(qh, kh, vh, cdec, ah);

  gemm_f16_kernel<false><<<gg, 256, 0, stream>>>(ah, Woh, (void*)out);
}
